// MultiHeadAttention_81544249081996
// MI455X (gfx1250) — matrix operand order repaired and checked by reading, not compiled, not run on hardware
//
#include <hip/hip_runtime.h>
#include <hip/hip_bf16.h>
#include <math.h>

// MHA forward for MI455X (gfx1250, wave32, WMMA).
// B=2, S=2048, HIDDEN=1024, HEADS=16, DK=64.
// Workspace layout (needs 56 MB):
//   [0]      Xq bf16 (8MB)  -- reused as Xattn after Q projection
//   [8MB]    Xk bf16
//   [16MB]   Xv bf16
//   [24MB]   Qp bf16 (scaled by 1/8)
//   [32MB]   Kp bf16
//   [40MB]   Vt bf16 (transposed: [b][h*64+d][s])
//   [48MB]   Wq/Wk/Wv/Wo bf16 (2MB each)

#define HID   1024
#define HEADS 16
#define DK    64
#define BATCH 2
#define SEQ   2048
#define NTOK  (BATCH*SEQ)

typedef __bf16 bf16_t;
typedef __attribute__((ext_vector_type(16))) __bf16 v16bf;
typedef __attribute__((ext_vector_type(8)))  __bf16 v8bf;
typedef __attribute__((ext_vector_type(4)))  __bf16 v4bf;
typedef __attribute__((ext_vector_type(8)))  float  v8f;

static __device__ __forceinline__ bf16_t f2bf(float f) {
  unsigned u = __float_as_uint(f);
  u += 0x7FFFu + ((u >> 16) & 1u);               // round-to-nearest-even
  unsigned short h = (unsigned short)(u >> 16);
  return __builtin_bit_cast(bf16_t, h);
}

static __device__ __forceinline__ v8f zero8() {
  v8f z;
  for (int i = 0; i < 8; i++) z[i] = 0.f;
  return z;
}

// Build a 16-element bf16 fragment from two contiguous 8-element (16B) loads
// located at p and p+gap.  gap=16 for A- and B-fragments (K runs 0..7 / 16..23).
static __device__ __forceinline__ v16bf ld_frag(const bf16_t* p, int gap) {
  v8bf lo = *(const v8bf*)p;
  v8bf hi = *(const v8bf*)(p + gap);
  return __builtin_shufflevector(lo, hi, 0,1,2,3,4,5,6,7,8,9,10,11,12,13,14,15);
}

static __device__ __forceinline__ v8f wmma_bf16(v16bf a, v16bf b, v8f c) {
  return __builtin_amdgcn_wmma_f32_16x16x32_bf16(false, a, false, b,
                                                 (short)0, c, false, false);
}

// Async global->LDS copy of 16 bytes (per-lane), tracked by ASYNCcnt.
// lds_off = LDS byte offset (low 32 bits of the generic shared address).
static __device__ __forceinline__ void async_ld_b128(unsigned lds_off,
                                                     const bf16_t* g) {
  unsigned long long ga = (unsigned long long)(const void*)g;
  asm volatile("global_load_async_to_lds_b128 %0, %1, off"
               :: "v"(lds_off), "v"(ga) : "memory");
}

// ---------------------------------------------------------------- convert
__global__ void cvt_f32_bf16(const float* __restrict__ x,
                             bf16_t* __restrict__ y, int n4) {
  int i = blockIdx.x * blockDim.x + threadIdx.x;
  if (i >= n4) return;
  float4 f = ((const float4*)x)[i];
  v4bf o;
  o[0] = f2bf(f.x); o[1] = f2bf(f.y); o[2] = f2bf(f.z); o[3] = f2bf(f.w);
  ((v4bf*)y)[i] = o;
}

// ---------------------------------------------------------------- GEMM
// C[M,1024] = A[M,1024] @ W[1024,1024]^T (+bias)*scale.
// Block = 128 threads (4 waves), block tile 128x128, wave tile 64x64.
// mode 0: bf16 row-major out    mode 1: fp32 row-major out
// mode 2: bf16 transposed out:  Vt[b][col][s]  (row = b*2048+s)
__global__ void __launch_bounds__(128)
gemm_wmma(const bf16_t* __restrict__ A, const bf16_t* __restrict__ W,
          const float* __restrict__ bias, float scale, int mode,
          bf16_t* __restrict__ out_bf, float* __restrict__ out_f) {
  const int lane = threadIdx.x & 31;
  const int w    = threadIdx.x >> 5;
  const int wm   = w >> 1, wn = w & 1;
  const int half = lane >> 4, l16 = lane & 15;
  const int R  = blockIdx.y * 128 + wm * 64;
  const int Cb = blockIdx.x * 128 + wn * 64;

  v8f acc[4][4];
  for (int a = 0; a < 4; a++)
    for (int b = 0; b < 4; b++) acc[a][b] = zero8();

  for (int kk = 0; kk < HID; kk += 32) {
    v16bf af[4], bfr[4];
#pragma unroll
    for (int mt = 0; mt < 4; mt++) {
      const bf16_t* p = A + (size_t)(R + mt*16 + l16) * HID + kk + half*8;
      af[mt] = ld_frag(p, 16);
    }
#pragma unroll
    for (int nt = 0; nt < 4; nt++) {
      const bf16_t* p = W + (size_t)(Cb + nt*16 + l16) * HID + kk + half*8;
      bfr[nt] = ld_frag(p, 16);
    }
#pragma unroll
    for (int mt = 0; mt < 4; mt++)
#pragma unroll
      for (int nt = 0; nt < 4; nt++)
        acc[mt][nt] = wmma_bf16(af[mt], bfr[nt], acc[mt][nt]);
  }

#pragma unroll
  for (int mt = 0; mt < 4; mt++)
#pragma unroll
    for (int nt = 0; nt < 4; nt++) {
      const int col = Cb + nt*16 + l16;
      const float bv = bias[col];
#pragma unroll
      for (int i = 0; i < 8; i++) {
        const int row = R + mt*16 + i + 8*half;
        const float v = (acc[mt][nt][i] + bv) * scale;
        if (mode == 0) {
          out_bf[(size_t)row * HID + col] = f2bf(v);
        } else if (mode == 1) {
          out_f[(size_t)row * HID + col] = v;
        } else {
          const int bb = row >> 11, s = row & (SEQ - 1);
          out_bf[((size_t)bb * HID + col) * SEQ + s] = f2bf(v);
        }
      }
    }
}

// ---------------------------------------------------------------- attention
// One wave = 16 query rows of one (b,h); block = 4 waves = 64 query rows.
// All 4 waves share the same 32-key chunk, so K/V chunks are staged into LDS
// once per block with GLOBAL_LOAD_ASYNC_TO_LDS_B128 (ASYNCcnt).  Staging is
// double-buffered: chunk i+1 is prefetched while chunk i is consumed, with a
// partial  s_wait_asynccnt 4  (async loads complete in order) to only wait on
// the chunk actually needed.  Qp already scaled by 1/sqrt(DK).
__global__ void __launch_bounds__(128)
flash_attn(const bf16_t* __restrict__ Qp, const bf16_t* __restrict__ Kp,
           const bf16_t* __restrict__ Vt, bf16_t* __restrict__ Xo) {
  __shared__ __align__(16) bf16_t ldsK[2][32][64];   // [buf][key][d]   8 KB
  __shared__ __align__(16) bf16_t ldsV[2][64][32];   // [buf][d][key]   8 KB
  __shared__ __align__(16) bf16_t pbuf[4][16][32];   // P staging       4 KB

  const int tid  = threadIdx.x;
  const int lane = tid & 31;
  const int w    = tid >> 5;
  const int half = lane >> 4, l16 = lane & 15;
  const int blk = blockIdx.x;
  const int qt = blk & 31;                 // S/64 = 32 query tiles
  const int h  = (blk >> 5) & (HEADS - 1);
  const int b  = blk >> 9;
  const int q0 = qt * 64 + w * 16;

  // LDS byte offsets (LDS aperture keeps the offset in addr[31:0])
  const unsigned offK = (unsigned)(size_t)&ldsK[0][0][0];
  const unsigned offV = (unsigned)(size_t)&ldsV[0][0][0];

  // per-thread staging segments (16B each): K = 32 rows x 8 segs,
  // V = 64 rows x 4 segs; 256 segments each, 2 per thread.
  const int kr0 = tid >> 3, kr1 = kr0 + 16, ks = tid & 7;
  const int vr0 = tid >> 2, vr1 = vr0 + 32, vs = tid & 3;

  // issue the 4 async 16B segment copies for key-chunk kc into buffer buf
  auto stage = [&](int kc, int buf) {
    const bf16_t* kg = Kp + (size_t)(b * SEQ + kc) * HID + h * DK;
    const bf16_t* vg = Vt + ((size_t)b * HID + h * DK) * SEQ + kc;
    const unsigned oK = offK + (unsigned)buf * (32u * 64u * 2u);
    const unsigned oV = offV + (unsigned)buf * (64u * 32u * 2u);
    async_ld_b128(oK + (unsigned)(kr0*64 + ks*8) * 2, kg + (size_t)kr0*HID + ks*8);
    async_ld_b128(oK + (unsigned)(kr1*64 + ks*8) * 2, kg + (size_t)kr1*HID + ks*8);
    async_ld_b128(oV + (unsigned)(vr0*32 + vs*8) * 2, vg + (size_t)vr0*SEQ + vs*8);
    async_ld_b128(oV + (unsigned)(vr1*32 + vs*8) * 2, vg + (size_t)vr1*SEQ + vs*8);
  };

  // Q A-fragments: rows q0+l16, d split 0..31 / 32..63
  const size_t qrow = (size_t)(b * SEQ + q0 + l16) * HID + h * DK;
  v16bf aq[2];
  aq[0] = ld_frag(Qp + qrow +  0 + half*8, 16);
  aq[1] = ld_frag(Qp + qrow + 32 + half*8, 16);

  v8f o[4];
  for (int i = 0; i < 4; i++) o[i] = zero8();
  float mst[8], lst[8];
  for (int i = 0; i < 8; i++) { mst[i] = -1e30f; lst[i] = 0.f; }

  stage(0, 0);                                // preload first chunk
  for (int kc = 0; kc < SEQ; kc += 32) {
    const int cur = (kc >> 5) & 1;
    if (kc + 32 < SEQ) {
      stage(kc + 32, cur ^ 1);                // prefetch next chunk
      asm volatile("s_wait_asynccnt 0x4" ::: "memory");  // current chunk landed
    } else {
      asm volatile("s_wait_asynccnt 0x0" ::: "memory");
    }
    __syncthreads();

    // ---- scores for 32 keys: two 16x16 C-tiles (key groups 0..15 / 16..31)
    v8f s0 = zero8(), s1 = zero8();
#pragma unroll
    for (int kf = 0; kf < 2; kf++) {
      s0 = wmma_bf16(aq[kf], ld_frag(&ldsK[cur][l16     ][kf*32 + half*8], 16), s0);
      s1 = wmma_bf16(aq[kf], ld_frag(&ldsK[cur][16 + l16][kf*32 + half*8], 16), s1);
    }

    // ---- online softmax; element i lives on row (i + 8*half)
    float alpha[8];
#pragma unroll
    for (int i = 0; i < 8; i++) {
      float t = fmaxf(s0[i], s1[i]);
      t = fmaxf(t, __shfl_xor(t, 1, 32));
      t = fmaxf(t, __shfl_xor(t, 2, 32));
      t = fmaxf(t, __shfl_xor(t, 4, 32));
      t = fmaxf(t, __shfl_xor(t, 8, 32));
      const float mn = fmaxf(mst[i], t);
      alpha[i] = __expf(mst[i] - mn);
      const float p0 = __expf(s0[i] - mn);
      const float p1 = __expf(s1[i] - mn);
      float rs = p0 + p1;
      rs += __shfl_xor(rs, 1, 32);
      rs += __shfl_xor(rs, 2, 32);
      rs += __shfl_xor(rs, 4, 32);
      rs += __shfl_xor(rs, 8, 32);
      lst[i] = lst[i] * alpha[i] + rs;
      mst[i] = mn;
      const int m = i + 8 * half;          // C-layout row
      pbuf[w][m][l16]      = f2bf(p0);     // -> row-major P[16][32] in LDS
      pbuf[w][m][16 + l16] = f2bf(p1);
    }
    asm volatile("s_wait_dscnt 0" ::: "memory");  // DS stores visible to re-read

#pragma unroll
    for (int dt = 0; dt < 4; dt++)
#pragma unroll
      for (int i = 0; i < 8; i++) o[dt][i] *= alpha[i];

    // P as A-fragment (16 rows x 32 keys) straight from LDS
    const v16bf ap = ld_frag(&pbuf[w][l16][half * 8], 16);

    // ---- PV: B-fragments from the staged V tile (keys contiguous per d-row)
#pragma unroll
    for (int dt = 0; dt < 4; dt++)
      o[dt] = wmma_bf16(ap, ld_frag(&ldsV[cur][dt*16 + l16][half*8], 16), o[dt]);

    __syncthreads();   // all reads of buf[cur] done before it is re-staged
  }

#pragma unroll
  for (int dt = 0; dt < 4; dt++)
#pragma unroll
    for (int i = 0; i < 8; i++) {
      const int m = i + 8 * half;
      const float val = o[dt][i] / lst[i];
      Xo[(size_t)(b*SEQ + q0 + m) * HID + h*DK + dt*16 + l16] = f2bf(val);
    }
}

// ---------------------------------------------------------------- launch
extern "C" void kernel_launch(void* const* d_in, const int* in_sizes, int n_in,
                              void* d_out, int out_size, void* d_ws, size_t ws_size,
                              hipStream_t stream) {
  const float* q  = (const float*)d_in[0];
  const float* k  = (const float*)d_in[1];
  const float* v  = (const float*)d_in[2];
  const float* Wq = (const float*)d_in[3];
  const float* bq = (const float*)d_in[4];
  const float* Wk = (const float*)d_in[5];
  const float* bk = (const float*)d_in[6];
  const float* Wv = (const float*)d_in[7];
  const float* bv = (const float*)d_in[8];
  const float* Wo = (const float*)d_in[9];
  const float* bo = (const float*)d_in[10];
  float* out = (float*)d_out;

  char* ws = (char*)d_ws;
  const size_t SZ_ACT = (size_t)NTOK * HID * sizeof(bf16_t);  // 8 MB
  const size_t SZ_W   = (size_t)HID * HID * sizeof(bf16_t);   // 2 MB
  bf16_t* Xq  = (bf16_t*)(ws);
  bf16_t* Xk  = (bf16_t*)(ws + 1 * SZ_ACT);
  bf16_t* Xv  = (bf16_t*)(ws + 2 * SZ_ACT);
  bf16_t* Qp  = (bf16_t*)(ws + 3 * SZ_ACT);
  bf16_t* Kp  = (bf16_t*)(ws + 4 * SZ_ACT);
  bf16_t* Vt  = (bf16_t*)(ws + 5 * SZ_ACT);
  bf16_t* Wqb = (bf16_t*)(ws + 6 * SZ_ACT);
  bf16_t* Wkb = (bf16_t*)(ws + 6 * SZ_ACT + 1 * SZ_W);
  bf16_t* Wvb = (bf16_t*)(ws + 6 * SZ_ACT + 2 * SZ_W);
  bf16_t* Wob = (bf16_t*)(ws + 6 * SZ_ACT + 3 * SZ_W);
  bf16_t* Xattn = Xq;  // Xq dead after Q projection; reuse

  const int n4a = NTOK * HID / 4;
  const int n4w = HID * HID / 4;
  cvt_f32_bf16<<<(n4a + 255) / 256, 256, 0, stream>>>(q, Xq, n4a);
  cvt_f32_bf16<<<(n4a + 255) / 256, 256, 0, stream>>>(k, Xk, n4a);
  cvt_f32_bf16<<<(n4a + 255) / 256, 256, 0, stream>>>(v, Xv, n4a);
  cvt_f32_bf16<<<(n4w + 255) / 256, 256, 0, stream>>>(Wq, Wqb, n4w);
  cvt_f32_bf16<<<(n4w + 255) / 256, 256, 0, stream>>>(Wk, Wkb, n4w);
  cvt_f32_bf16<<<(n4w + 255) / 256, 256, 0, stream>>>(Wv, Wvb, n4w);
  cvt_f32_bf16<<<(n4w + 255) / 256, 256, 0, stream>>>(Wo, Wob, n4w);

  dim3 gg(HID / 128, NTOK / 128);   // (8, 32)
  dim3 gb(128);
  gemm_wmma<<<gg, gb, 0, stream>>>(Xq, Wqb, bq, 0.125f, 0, Qp, nullptr);
  gemm_wmma<<<gg, gb, 0, stream>>>(Xk, Wkb, bk, 1.0f,   0, Kp, nullptr);
  gemm_wmma<<<gg, gb, 0, stream>>>(Xv, Wvb, bv, 1.0f,   2, Vt, nullptr);

  flash_attn<<<dim3(BATCH * HEADS * (SEQ / 64)), gb, 0, stream>>>(Qp, Kp, Vt, Xattn);

  gemm_wmma<<<gg, gb, 0, stream>>>(Xattn, Wob, bo, 1.0f, 1, nullptr, out);
}
